// StrongGNNModel_223338299460
// MI455X (gfx1250) — matrix operand order repaired and checked by reading, not compiled, not run on hardware
//
#include <hip/hip_runtime.h>
#include <hip/hip_bf16.h>

// ---------------------------------------------------------------------------
// Problem constants (match reference)
// ---------------------------------------------------------------------------
#define NN 200000
#define EDGES 800000
#define BB 4096
#define HH 256
#define EEDIM 128
#define LL 5
#define EXP_IN 64
#define EXP_H 256
#define COMB_H 512
#define COMB_IN (2*HH + EXP_H)   // 768
#define BN_EPS 1e-5f

typedef unsigned short u16;
typedef __attribute__((ext_vector_type(16))) u16     v16u;
typedef __attribute__((ext_vector_type(8)))  u16     v8u;
typedef __attribute__((ext_vector_type(16))) __bf16  v16bf;
typedef __attribute__((ext_vector_type(8)))  float   v8f;
typedef int v4i __attribute__((vector_size(16)));    // matches builtin param type

// CDNA5 async global->LDS DMA (ASYNCcnt-tracked), guarded for toolchain support
#if __has_builtin(__builtin_amdgcn_global_load_async_to_lds_b128)
#define HAS_ASYNC_LDS 1
#else
#define HAS_ASYNC_LDS 0
#endif

__device__ __forceinline__ void wait_async0() {
#if __has_builtin(__builtin_amdgcn_s_wait_asynccnt)
    __builtin_amdgcn_s_wait_asynccnt(0);
#else
    asm volatile("s_wait_asynccnt 0" ::: "memory");
#endif
}

// float -> bf16 with round-to-nearest-even
__device__ __forceinline__ u16 tobf(float f) {
    unsigned u = __float_as_uint(f);
    u += 0x7fffu + ((u >> 16) & 1u);
    return (u16)(u >> 16);
}
// monotone key encoding of float for atomicMax on unsigned
__device__ __forceinline__ unsigned fkey(float f) {
    unsigned u = __float_as_uint(f);
    return (u & 0x80000000u) ? ~u : (u | 0x80000000u);
}
__device__ __forceinline__ float fdecode(unsigned k) {
    unsigned u = (k & 0x80000000u) ? (k & 0x7fffffffu) : ~k;
    return __uint_as_float(u);
}

// ---------------------------------------------------------------------------
// Small helper kernels
// ---------------------------------------------------------------------------
__global__ void k_cvt_bf16(const float* __restrict__ s, u16* __restrict__ d, int n) {
    int i = blockIdx.x * 256 + threadIdx.x;
    if (i < n) d[i] = tobf(s[i]);
}

// W [K x N] f32  ->  WT [N x K] bf16   (one-time; keeps GEMM B-tiles K-contiguous)
__global__ void k_cvt_bf16_t(const float* __restrict__ s, u16* __restrict__ d, int K, int N) {
    int i = blockIdx.x * 256 + threadIdx.x;
    if (i >= K * N) return;
    int k = i / N, n = i - k * N;
    d[(size_t)n * K + k] = tobf(s[i]);
}

// h0[n,c] = atom_emb[clip(x0)][c] + chiral_emb[clip(x1)][c]
__global__ void k_init_nodes(const int* __restrict__ xn, const float* __restrict__ aemb,
                             const float* __restrict__ cemb, float* __restrict__ h, int n) {
    int i = blockIdx.x * 256 + threadIdx.x;
    if (i >= n) return;
    int node = i >> 8, c = i & 255;
    int a = xn[node * 2];     a = a < 0 ? 0 : (a > 119 ? 119 : a);
    int ch = xn[node * 2 + 1]; ch = ch < 0 ? 0 : (ch > 3 ? 3 : ch);
    h[i] = aemb[a * HH + c] + cemb[ch * HH + c];
}

// tab[(l*24 + t*4+d), o] = (edge_type_emb[t]+edge_dir_emb[d]) @ edge_lin_W[l] + edge_lin_b[l]
__global__ void k_table(const float* __restrict__ te, const float* __restrict__ de,
                        const float* __restrict__ lw, const float* __restrict__ lb,
                        float* __restrict__ tab) {
    int bidx = blockIdx.x;          // l*24 + combo
    int l = bidx / 24, c = bidx % 24;
    int t = c / 4, d = c % 4;
    int o = threadIdx.x;            // 0..255
    float acc = lb[l * HH + o];
    const float* tpe = te + t * EEDIM;
    const float* dpe = de + d * EEDIM;
    const float* w = lw + (size_t)l * EEDIM * HH;
    #pragma unroll 4
    for (int k = 0; k < EEDIM; ++k)
        acc += (tpe[k] + dpe[k]) * w[k * HH + o];
    tab[(size_t)bidx * HH + o] = acc;
}

// fused BN constants: scale = gamma*rsqrt(var+eps), shift = beta - mean*scale
__global__ void k_bnprep(const float* __restrict__ g, const float* __restrict__ b,
                         const float* __restrict__ m, const float* __restrict__ v,
                         float* __restrict__ S, float* __restrict__ T, int n) {
    int i = blockIdx.x * 256 + threadIdx.x;
    if (i >= n) return;
    float s = g[i] * rsqrtf(v[i] + BN_EPS);
    S[i] = s;
    T[i] = b[i] - m[i] * s;
}

// message + scatter: one wave per edge, 8 floats per lane (H=256)
__global__ __launch_bounds__(256) void k_msg(const int* __restrict__ ei, const int* __restrict__ ea,
                                             const float* __restrict__ h, const float* __restrict__ tab,
                                             float* __restrict__ agg, int ne) {
    int gw = blockIdx.x * 8 + (threadIdx.x >> 5);
    if (gw >= ne) return;
    int lane = threadIdx.x & 31;
    int src = ei[gw];
    int dst = ei[ne + gw];
    int t = ea[gw * 2];     t = t < 0 ? 0 : (t > 5 ? 5 : t);
    int d = ea[gw * 2 + 1]; d = d < 0 ? 0 : (d > 3 ? 3 : d);
    const float* tp = tab + (t * 4 + d) * HH + lane * 8;
    const float* hp = h + (size_t)src * HH + lane * 8;
    float* ap = agg + (size_t)dst * HH + lane * 8;
    #pragma unroll
    for (int j = 0; j < 8; ++j) {
        float v = fmaxf(hp[j] + tp[j], 0.0f);
        atomicAdd(&ap[j], v);
    }
}

// zA = bf16(h + agg)
__global__ void k_prepz(const float* __restrict__ h, const float* __restrict__ agg,
                        u16* __restrict__ z, int n) {
    int i = blockIdx.x * 256 + threadIdx.x;
    if (i < n) z[i] = tobf(h[i] + agg[i]);
}

// ---------------------------------------------------------------------------
// WMMA GEMM: D = epilogue(A[M,K](bf16, row-major) @ WT[N,K](bf16)^T + bias)
// 64x64 tile per 256-thread block (8 waves, each: 1 A-strip x 2 16x16 tiles).
// K staged 32 at a time through double-buffered LDS, one barrier per k-step.
//  - async path: GLOBAL_LOAD_ASYNC_TO_LDS_B128 DMA for next buffer issued
//    before the WMMAs; s_wait_asynccnt 0 + barrier before consuming.
//  - fallback:   branch-free register pipeline (loads and LDS stores are
//    unconditional so the scheduler cannot sink them past the WMMAs).
// EPI: 1 = relu (write f32 outF and/or bf16 outB); 2 = BN+relu+residual into hres
// ---------------------------------------------------------------------------
#define LDT_S 40   // 32 K-elems + 8 pad (u16); row pitch 80B (16B aligned)
template<int EPI>
__global__ __launch_bounds__(256) void k_gemm(const u16* __restrict__ A, int lda,
                                              const u16* __restrict__ WT, int ldw,
                                              const float* __restrict__ bias,
                                              float* __restrict__ outF, int ldF,
                                              u16* __restrict__ outB, int ldB,
                                              const float* __restrict__ bnS,
                                              const float* __restrict__ bnT,
                                              float* __restrict__ hres,
                                              int K) {
    __shared__ __align__(16) u16 sA[2][64 * LDT_S];   // [m][k]
    __shared__ __align__(16) u16 sB[2][64 * LDT_S];   // [n][k]

    const int m0 = blockIdx.x * 64;
    const int n0 = blockIdx.y * 64;
    const int tid = threadIdx.x;
    const int lane = tid & 31;
    const int wave = tid >> 5;
    const int tm = wave & 3;          // tile row (M)
    const int tn0 = (wave >> 2) * 2;  // first of two tile cols (N)

    const int sr = tid >> 2, sc = (tid & 3) * 8;   // staging: 64 rows x 32 cols, 8 u16/thread
    const int ml = tm * 16 + (lane & 15);
    const int half = lane >> 4;

    const u16* Ap = A  + (size_t)(m0 + sr) * lda + sc;
    const u16* Wp = WT + (size_t)(n0 + sr) * ldw + sc;
    const int sidx = sr * LDT_S + sc;

    v8f zero = {0.f, 0.f, 0.f, 0.f, 0.f, 0.f, 0.f, 0.f};
    v8f acc[2]; acc[0] = zero; acc[1] = zero;

    // fragment compute for one staged k-step
    auto compute = [&](int cb) {
        // A fragment 16x32 bf16: lanes 0-15 own K 0-7 & 16-23; lanes 16-31 K 8-15 & 24-31
        v8u a0 = *(const v8u*)&sA[cb][ml * LDT_S + half * 8];
        v8u a1 = *(const v8u*)&sA[cb][ml * LDT_S + 16 + half * 8];
        v16u au = __builtin_shufflevector(a0, a1, 0, 1, 2, 3, 4, 5, 6, 7,
                                          8, 9, 10, 11, 12, 13, 14, 15);
        v16bf av = __builtin_bit_cast(v16bf, au);
        #pragma unroll
        for (int t = 0; t < 2; ++t) {
            // B fragment 32x16 bf16: lanes 0-15 own K 0-7 & 16-23; lanes 16-31 K 8-15 & 24-31 for column N=lane&15
            const int nl = (tn0 + t) * 16 + (lane & 15);
            v8u b0 = *(const v8u*)&sB[cb][nl * LDT_S + half * 8];
            v8u b1 = *(const v8u*)&sB[cb][nl * LDT_S + 16 + half * 8];
            v16u bu = __builtin_shufflevector(b0, b1, 0, 1, 2, 3, 4, 5, 6, 7,
                                              8, 9, 10, 11, 12, 13, 14, 15);
            v16bf bv = __builtin_bit_cast(v16bf, bu);
            acc[t] = __builtin_amdgcn_wmma_f32_16x16x32_bf16(
                false, av, false, bv, (short)0, acc[t], false, false);
        }
    };

#if HAS_ASYNC_LDS
    // ---- async DMA double-buffer pipeline (ASYNCcnt) ----
    // builtin signature: (v4i addrspace(1)*, v4i addrspace(3)*, imm offset, imm cpol)
    #define GPTR(p) ((__attribute__((address_space(1))) v4i*)(p))
    #define LPTR(p) ((__attribute__((address_space(3))) v4i*)(p))
    __builtin_amdgcn_global_load_async_to_lds_b128(GPTR(Ap), LPTR(&sA[0][sidx]), 0, 0);
    __builtin_amdgcn_global_load_async_to_lds_b128(GPTR(Wp), LPTR(&sB[0][sidx]), 0, 0);
    wait_async0();
    __syncthreads();
    int cur = 0;
    for (int k0 = 0; k0 < K; k0 += 32, cur ^= 1) {
        // kick off DMA for the next k-step (clamped on last iter; dead buffer)
        const int kn = (k0 + 32 < K) ? (k0 + 32) : 0;
        __builtin_amdgcn_global_load_async_to_lds_b128(GPTR(Ap + kn), LPTR(&sA[cur ^ 1][sidx]), 0, 0);
        __builtin_amdgcn_global_load_async_to_lds_b128(GPTR(Wp + kn), LPTR(&sB[cur ^ 1][sidx]), 0, 0);
        compute(cur);          // WMMAs overlap the in-flight DMA
        wait_async0();         // this wave's DMA writes have landed in LDS
        __syncthreads();       // all waves' DMA + reads of buf[cur] complete
    }
    #undef GPTR
    #undef LPTR
#else
    // ---- register double-buffer pipeline (branch-free body) ----
    *(v8u*)&sA[0][sidx] = *(const v8u*)(Ap);
    *(v8u*)&sB[0][sidx] = *(const v8u*)(Wp);
    int cur = 0;
    for (int k0 = 0; k0 < K; k0 += 32, cur ^= 1) {
        __syncthreads();   // buf[cur] stores visible; buf[cur^1] readers done
        const int kn = (k0 + 32 < K) ? (k0 + 32) : 0;
        v8u ra = *(const v8u*)(Ap + kn);
        v8u rb = *(const v8u*)(Wp + kn);
        compute(cur);
        // unconditional: last iteration writes the dead buffer (never read)
        *(v8u*)&sA[cur ^ 1][sidx] = ra;
        *(v8u*)&sB[cur ^ 1][sidx] = rb;
    }
#endif

    // Epilogue (C/D layout: VGPR v -> M = v + 8*half, lanes&15 -> N)
    #pragma unroll
    for (int t = 0; t < 2; ++t) {
        const int col = n0 + (tn0 + t) * 16 + (lane & 15);
        const float bi = bias ? bias[col] : 0.0f;
        #pragma unroll
        for (int v = 0; v < 8; ++v) {
            const int row = m0 + tm * 16 + v + 8 * half;
            float x = acc[t][v] + bi;
            if (EPI == 2) {
                x = x * bnS[col] + bnT[col];
                size_t idx = (size_t)row * ldF + col;
                hres[idx] = fmaxf(x, 0.0f) + hres[idx];
            } else {
                if (EPI == 1) x = fmaxf(x, 0.0f);
                if (outF) outF[(size_t)row * ldF + col] = x;
                if (outB) outB[(size_t)row * ldB + col] = tobf(x);
            }
        }
    }
}

// ---------------------------------------------------------------------------
// Pooling
// ---------------------------------------------------------------------------
__global__ __launch_bounds__(256) void k_pool(const int* __restrict__ batch, const float* __restrict__ h,
                                              float* __restrict__ psum, unsigned* __restrict__ pmax,
                                              float* __restrict__ pcnt, int n) {
    int gw = blockIdx.x * 8 + (threadIdx.x >> 5);
    if (gw >= n) return;
    int lane = threadIdx.x & 31;
    int b = batch[gw];
    const float* hp = h + (size_t)gw * HH + lane * 8;
    float* sp = psum + (size_t)b * HH + lane * 8;
    unsigned* mp = pmax + (size_t)b * HH + lane * 8;
    #pragma unroll
    for (int j = 0; j < 8; ++j) {
        float v = hp[j];
        atomicAdd(&sp[j], v);
        atomicMax(&mp[j], fkey(v));
    }
    if (lane == 0) atomicAdd(&pcnt[b], 1.0f);
}

__global__ void k_finalize(const float* __restrict__ psum, const unsigned* __restrict__ pmax,
                           const float* __restrict__ pcnt, float* __restrict__ dout,
                           u16* __restrict__ combb) {
    int i = blockIdx.x * 256 + threadIdx.x;
    if (i >= BB * HH) return;
    int b = i >> 8, c = i & 255;
    float cnt = fmaxf(pcnt[b], 1.0f);
    float mean = psum[i] / cnt;
    float mx = fdecode(pmax[i]);
    if (!isfinite(mx)) mx = 0.0f;
    const size_t GOFF = BB;
    const size_t COFF = (size_t)BB + (size_t)BB * (2 * HH);
    dout[GOFF + (size_t)b * (2 * HH) + c] = mean;
    dout[GOFF + (size_t)b * (2 * HH) + HH + c] = mx;
    dout[COFF + (size_t)b * COMB_IN + c] = mean;
    dout[COFF + (size_t)b * COMB_IN + HH + c] = mx;
    combb[(size_t)b * COMB_IN + c] = tobf(mean);
    combb[(size_t)b * COMB_IN + HH + c] = tobf(mx);
}

// final 256 -> 1 projection, wave per row
__global__ __launch_bounds__(256) void k_final_dot(const float* __restrict__ y2,
                                                   const float* __restrict__ w3,
                                                   const float* __restrict__ b3,
                                                   float* __restrict__ dout) {
    int b = blockIdx.x * 8 + (threadIdx.x >> 5);
    if (b >= BB) return;
    int lane = threadIdx.x & 31;
    float s = 0.0f;
    #pragma unroll
    for (int j = 0; j < 8; ++j)
        s += y2[(size_t)b * 256 + j * 32 + lane] * w3[j * 32 + lane];
    #pragma unroll
    for (int off = 16; off > 0; off >>= 1)
        s += __shfl_xor(s, off, 32);
    if (lane == 0) dout[b] = s + b3[0];
}

// ---------------------------------------------------------------------------
// Launcher
// ---------------------------------------------------------------------------
extern "C" void kernel_launch(void* const* d_in, const int* in_sizes, int n_in,
                              void* d_out, int out_size, void* d_ws, size_t ws_size,
                              hipStream_t stream) {
    (void)in_sizes; (void)n_in; (void)out_size; (void)ws_size;

    const int*   x_nodes  = (const int*)d_in[0];
    const int*   ei       = (const int*)d_in[1];
    const int*   ea       = (const int*)d_in[2];
    const int*   batch    = (const int*)d_in[3];
    const float* expfeat  = (const float*)d_in[4];
    const float* atom_emb = (const float*)d_in[5];
    const float* chir_emb = (const float*)d_in[6];
    const float* te_emb   = (const float*)d_in[7];
    const float* de_emb   = (const float*)d_in[8];
    const float* elW      = (const float*)d_in[9];
    const float* elB      = (const float*)d_in[10];
    const float* mW1      = (const float*)d_in[11];
    const float* mB1      = (const float*)d_in[12];
    const float* mW2      = (const float*)d_in[13];
    const float* mB2      = (const float*)d_in[14];
    const float* bnG      = (const float*)d_in[15];
    const float* bnB      = (const float*)d_in[16];
    const float* bnM      = (const float*)d_in[17];
    const float* bnV      = (const float*)d_in[18];
    const float* eW1      = (const float*)d_in[19];
    const float* eB1      = (const float*)d_in[20];
    const float* eW2      = (const float*)d_in[21];
    const float* eB2      = (const float*)d_in[22];
    const float* hW1      = (const float*)d_in[23];
    const float* hB1      = (const float*)d_in[24];
    const float* hW2      = (const float*)d_in[25];
    const float* hB2      = (const float*)d_in[26];
    const float* hW3      = (const float*)d_in[27];
    const float* hB3      = (const float*)d_in[28];
    float* dout = (float*)d_out;

    // workspace carve-up (256B aligned)
    char* p = (char*)d_ws;
    auto alloc = [&](size_t bytes) -> void* {
        void* r = (void*)p;
        p += (bytes + 255) & ~(size_t)255;
        return r;
    };
    float*    h      = (float*)alloc((size_t)NN * HH * 4);
    float*    agg    = (float*)alloc((size_t)NN * HH * 4);
    u16*      zA     = (u16*)  alloc((size_t)NN * HH * 2);
    u16*      z1     = (u16*)  alloc((size_t)NN * HH * 2);
    float*    tab    = (float*)alloc((size_t)LL * 24 * HH * 4);
    float*    bnS    = (float*)alloc((size_t)LL * HH * 4);
    float*    bnT    = (float*)alloc((size_t)LL * HH * 4);
    u16*      W1b    = (u16*)  alloc((size_t)LL * HH * HH * 2);   // transposed per layer
    u16*      W2b    = (u16*)  alloc((size_t)LL * HH * HH * 2);   // transposed per layer
    u16*      eW1b   = (u16*)  alloc((size_t)EXP_IN * EXP_H * 2); // [256][64]
    u16*      eW2b   = (u16*)  alloc((size_t)EXP_H * EXP_H * 2);  // [256][256]
    u16*      hW1b   = (u16*)  alloc((size_t)COMB_IN * COMB_H * 2); // [512][768]
    u16*      hW2b   = (u16*)  alloc((size_t)COMB_H * (COMB_H / 2) * 2); // [256][512]
    u16*      expfb  = (u16*)  alloc((size_t)BB * EXP_IN * 2);
    u16*      exp1b  = (u16*)  alloc((size_t)BB * EXP_H * 2);
    u16*      combb  = (u16*)  alloc((size_t)BB * COMB_IN * 2);
    u16*      y1b    = (u16*)  alloc((size_t)BB * COMB_H * 2);
    float*    y2     = (float*)alloc((size_t)BB * (COMB_H / 2) * 4);
    float*    psum   = (float*)alloc((size_t)BB * HH * 4);
    unsigned* pmax   = (unsigned*)alloc((size_t)BB * HH * 4);
    float*    pcnt   = (float*)alloc((size_t)BB * 4);

    auto blocks = [](size_t n) { return (unsigned)((n + 255) / 256); };

    // weight conversions to transposed bf16 ([N][K])
    for (int l = 0; l < LL; ++l) {
        k_cvt_bf16_t<<<blocks((size_t)HH * HH), 256, 0, stream>>>(
            mW1 + (size_t)l * HH * HH, W1b + (size_t)l * HH * HH, HH, HH);
        k_cvt_bf16_t<<<blocks((size_t)HH * HH), 256, 0, stream>>>(
            mW2 + (size_t)l * HH * HH, W2b + (size_t)l * HH * HH, HH, HH);
    }
    k_cvt_bf16_t<<<blocks((size_t)EXP_IN * EXP_H), 256, 0, stream>>>(eW1, eW1b, EXP_IN, EXP_H);
    k_cvt_bf16_t<<<blocks((size_t)EXP_H * EXP_H), 256, 0, stream>>>(eW2, eW2b, EXP_H, EXP_H);
    k_cvt_bf16_t<<<blocks((size_t)COMB_IN * COMB_H), 256, 0, stream>>>(hW1, hW1b, COMB_IN, COMB_H);
    k_cvt_bf16_t<<<blocks((size_t)COMB_H * (COMB_H / 2)), 256, 0, stream>>>(hW2, hW2b, COMB_H, COMB_H / 2);
    // activations (row-major bf16)
    k_cvt_bf16<<<blocks((size_t)BB * EXP_IN), 256, 0, stream>>>(expfeat, expfb, BB * EXP_IN);

    // node init, edge-projection table (24 distinct edge embeddings!), BN constants
    k_init_nodes<<<blocks((size_t)NN * HH), 256, 0, stream>>>(x_nodes, atom_emb, chir_emb, h, NN * HH);
    k_table<<<LL * 24, 256, 0, stream>>>(te_emb, de_emb, elW, elB, tab);
    k_bnprep<<<blocks(LL * HH), 256, 0, stream>>>(bnG, bnB, bnM, bnV, bnS, bnT, LL * HH);

    // GINE layers
    for (int l = 0; l < LL; ++l) {
        (void)hipMemsetAsync(agg, 0, (size_t)NN * HH * 4, stream);
        k_msg<<<EDGES / 8, 256, 0, stream>>>(ei, ea, h, tab + (size_t)l * 24 * HH, agg, EDGES);
        k_prepz<<<blocks((size_t)NN * HH), 256, 0, stream>>>(h, agg, zA, NN * HH);
        dim3 g1(NN / 64, HH / 64);
        k_gemm<1><<<g1, 256, 0, stream>>>(zA, HH, W1b + (size_t)l * HH * HH, HH, mB1 + l * HH,
                                          nullptr, 0, z1, HH, nullptr, nullptr, nullptr, HH);
        k_gemm<2><<<g1, 256, 0, stream>>>(z1, HH, W2b + (size_t)l * HH * HH, HH, mB2 + l * HH,
                                          nullptr, HH, nullptr, 0,
                                          bnS + l * HH, bnT + l * HH, h, HH);
    }

    // readout
    (void)hipMemsetAsync(psum, 0, (size_t)BB * HH * 4, stream);
    (void)hipMemsetAsync(pmax, 0, (size_t)BB * HH * 4, stream);
    (void)hipMemsetAsync(pcnt, 0, (size_t)BB * 4, stream);
    k_pool<<<NN / 8, 256, 0, stream>>>(batch, h, psum, pmax, pcnt, NN);
    k_finalize<<<blocks((size_t)BB * HH), 256, 0, stream>>>(psum, pmax, pcnt, dout, combb);

    // experimental-feature MLP
    const size_t COFF = (size_t)BB + (size_t)BB * (2 * HH);
    k_gemm<1><<<dim3(BB / 64, EXP_H / 64), 256, 0, stream>>>(expfb, EXP_IN, eW1b, EXP_IN, eB1,
                                                             nullptr, 0, exp1b, EXP_H,
                                                             nullptr, nullptr, nullptr, EXP_IN);
    k_gemm<1><<<dim3(BB / 64, EXP_H / 64), 256, 0, stream>>>(exp1b, EXP_H, eW2b, EXP_H, eB2,
                                                             dout + COFF + 2 * HH, COMB_IN,
                                                             combb + 2 * HH, COMB_IN,
                                                             nullptr, nullptr, nullptr, EXP_H);

    // head
    k_gemm<1><<<dim3(BB / 64, COMB_H / 64), 256, 0, stream>>>(combb, COMB_IN, hW1b, COMB_IN, hB1,
                                                              nullptr, 0, y1b, COMB_H,
                                                              nullptr, nullptr, nullptr, COMB_IN);
    k_gemm<1><<<dim3(BB / 64, (COMB_H / 2) / 64), 256, 0, stream>>>(y1b, COMB_H, hW2b, COMB_H, hB2,
                                                                    y2, COMB_H / 2, nullptr, 0,
                                                                    nullptr, nullptr, nullptr, COMB_H);
    k_final_dot<<<BB / 8, 256, 0, stream>>>(y2, hW3, hB3, dout);
}
